// RecurrentNeuralNetwork_58420145160468
// MI455X (gfx1250) — compile-verified, matrix operand order checked by reading, not run on hardware
//
#include <hip/hip_runtime.h>
#include <hip/hip_bf16.h>

typedef __bf16 bf16_t;
typedef __attribute__((ext_vector_type(8)))  __bf16 v8bf;
typedef __attribute__((ext_vector_type(16))) __bf16 v16bf;
typedef __attribute__((ext_vector_type(8)))  float  v8f;

#define T_STEPS 256
#define NB      128
#define NIN     128
#define NHID    1024
#define NOUT    128
#define KTOT    (NIN + NHID)   // 1152

// d_out layout (floats): hidden_list [128][256][1024], output_list [128][256][128], h_final [128][1024]
#define HID_STRIDE ((size_t)T_STEPS * NHID)
#define OUT_OFF    ((size_t)NB * T_STEPS * NHID)            // 33,554,432
#define HF_OFF     (OUT_OFF + (size_t)NB * T_STEPS * NOUT)  // 37,748,736

// ws layout (bytes)
#define WCAT_OFF 0u                                   // bf16 [NHID][KTOT]
#define WOUT_OFF (WCAT_OFF + NHID * KTOT * 2u)        // bf16 [NOUT][NHID]
#define XCAT_OFF (WOUT_OFF + NOUT * NHID * 2u)        // bf16 [T][NB][NIN]
#define BN_OFF   (XCAT_OFF + T_STEPS * NB * NIN * 2u) // f32  [T][NHID]
#define ABUF_OFF (BN_OFF + T_STEPS * NHID * 4u)       // bf16 2x[NB][KTOT]

// Load a 16-element bf16 fragment: two contiguous 8-elem (16B) chunks at +0 and +16,
// matching the 16-bit A/B VGPR layout for K=32 (per-lane K-chunks base and base+16).
__device__ __forceinline__ v16bf load_frag(const bf16_t* p) {
    v8bf lo = *(const v8bf*)(p);
    v8bf hi = *(const v8bf*)(p + 16);
    return __builtin_shufflevector(lo, hi, 0,1,2,3,4,5,6,7,8,9,10,11,12,13,14,15);
}

// Row-pointer GEMM core: accumulate K columns (multiple of 32) into a 16x16 f32 tile.
__device__ __forceinline__ v8f wmma_k_loop(const bf16_t* arow, const bf16_t* brow, int K) {
    v8f c = {};
    #pragma unroll 4
    for (int k0 = 0; k0 < K; k0 += 32) {
        v16bf a = load_frag(arow + k0);
        v16bf b = load_frag(brow + k0);
        c = __builtin_amdgcn_wmma_f32_16x16x32_bf16(
                false, a, false, b, (short)0, c, false, false);
    }
    return c;
}

// ---------------- prologue: convert weights/inputs to bf16, precompute bias+noise ----
__global__ void rnn_prep(const float* __restrict__ x,       // [NB][T][NIN]
                         const float* __restrict__ hidden,  // [NB][NHID]
                         const float* __restrict__ W_in,    // [NHID][NIN]
                         const float* __restrict__ W_hh,    // [NHID][NHID]
                         const float* __restrict__ b_hh,    // [NHID]
                         const float* __restrict__ W_out,   // [NOUT][NHID]
                         const float* __restrict__ alpha_w, // [NHID]
                         const float* __restrict__ noise,   // [T][NHID]
                         bf16_t* __restrict__ Wcat,
                         bf16_t* __restrict__ Wout,
                         bf16_t* __restrict__ xcat,
                         float*  __restrict__ bias_noise,
                         bf16_t* __restrict__ abuf0) {
    const int N0 = NHID * KTOT;
    const int N1 = NOUT * NHID;
    const int N2 = T_STEPS * NB * NIN;
    const int N3 = T_STEPS * NHID;
    const int N4 = NB * NIN;
    const int N5 = NB * NHID;
    const int total = N0 + N1 + N2 + N3 + N4 + N5;
    for (int i = blockIdx.x * blockDim.x + threadIdx.x; i < total;
         i += gridDim.x * blockDim.x) {
        int idx = i;
        if (idx < N0) {
            int j = idx / KTOT, k = idx % KTOT;
            float w = (k < NIN) ? W_in[j * NIN + k] : W_hh[j * NHID + (k - NIN)];
            Wcat[idx] = (bf16_t)w;
            continue;
        }
        idx -= N0;
        if (idx < N1) { Wout[idx] = (bf16_t)W_out[idx]; continue; }
        idx -= N1;
        if (idx < N2) {
            int t = idx / (NB * NIN);
            int rem = idx % (NB * NIN);
            int b = rem / NIN, k = rem % NIN;
            xcat[idx] = (bf16_t)x[(size_t)b * (T_STEPS * NIN) + t * NIN + k];
            continue;
        }
        idx -= N2;
        if (idx < N3) {
            int j = idx % NHID;
            bias_noise[idx] = b_hh[j] + noise[idx] * sqrtf(2.0f / alpha_w[j]) * 0.05f;
            continue;
        }
        idx -= N3;
        if (idx < N4) {
            int b = idx / NIN, k = idx % NIN;
            abuf0[b * KTOT + k] = (bf16_t)x[(size_t)b * (T_STEPS * NIN) + k]; // t = 0
            continue;
        }
        idx -= N4;
        {
            int b = idx / NHID, j = idx % NHID;
            abuf0[b * KTOT + NIN + j] = (bf16_t)hidden[idx];
        }
    }
}

// ---------------- fused per-step kernel ---------------------------------------------
// Blocks 0..63  : hidden GEMM h_t = relu([x_t|h_{t-1}] @ Wcat^T + bias_noise[t])
//                 (512 wave-tiles: 8 mb x 64 jb; writes hidden_list, h_final, abuf_w h-part)
// Blocks 64..71 : output GEMM y_{t-1} = h_{t-1} @ W_out^T  (64 wave-tiles; reads abuf_r)
//                 + stage x_{t+1} bf16 into abuf_w x-part (read by nothing this launch)
__global__ __launch_bounds__(256) void rnn_fused_step(
        const bf16_t* __restrict__ abuf_r,    // [NB][KTOT]  x_t | h_{t-1}
        bf16_t* __restrict__ abuf_w,          // [NB][KTOT]  receives h_t and x_{t+1}
        const bf16_t* __restrict__ Wcat,      // [NHID][KTOT]
        const bf16_t* __restrict__ Wout,      // [NOUT][NHID]
        const bf16_t* __restrict__ xcat,      // [T][NB][NIN]
        const float* __restrict__ bias_noise, // [T][NHID]
        float* __restrict__ out_base,         // d_out
        int t) {
    const int lane = threadIdx.x & 31;
    const int half = lane >> 4;
    const int q    = lane & 15;

    if (blockIdx.x < 64) {
        // ---- hidden-state GEMM ----
        const int wave = blockIdx.x * 8 + (threadIdx.x >> 5);
        const int mb = wave >> 6;   // 0..7
        const int jb = wave & 63;   // 0..63

        const bf16_t* arow = abuf_r + (size_t)(mb * 16 + q) * KTOT + half * 8;
        const bf16_t* brow = Wcat   + (size_t)(jb * 16 + q) * KTOT + half * 8;
        v8f c = wmma_k_loop(arow, brow, KTOT);

        const int j = jb * 16 + q;
        const float bn = bias_noise[t * NHID + j];
        #pragma unroll
        for (int r = 0; r < 8; ++r) {
            const int b = mb * 16 + half * 8 + r;   // C layout: VGPR r -> row r + 8*half
            float v = fmaxf(c[r] + bn, 0.0f);
            out_base[(size_t)b * HID_STRIDE + (size_t)t * NHID + j] = v;
            if (t == T_STEPS - 1)
                out_base[HF_OFF + (size_t)b * NHID + j] = v;
            abuf_w[(size_t)b * KTOT + NIN + j] = (bf16_t)v;
        }
    } else {
        // ---- output GEMM for the PREVIOUS step (h_{t-1} lives in abuf_r) ----
        if (t > 0) {
            const int wave = (blockIdx.x - 64) * 8 + (threadIdx.x >> 5); // 0..63
            const int mb = wave >> 3;   // 0..7
            const int ob = wave & 7;    // 0..7

            const bf16_t* arow = abuf_r + (size_t)(mb * 16 + q) * KTOT + NIN + half * 8;
            const bf16_t* brow = Wout   + (size_t)(ob * 16 + q) * NHID + half * 8;
            v8f c = wmma_k_loop(arow, brow, NHID);

            const int o = ob * 16 + q;
            float* out_y = out_base + OUT_OFF;
            #pragma unroll
            for (int r = 0; r < 8; ++r) {
                const int b = mb * 16 + half * 8 + r;
                out_y[(size_t)b * (T_STEPS * NOUT) + (size_t)(t - 1) * NOUT + o] = c[r];
            }
        }
        // ---- stage x_{t+1} into abuf_w x-part ----
        if (t < T_STEPS - 1) {
            const int tid2 = (blockIdx.x - 64) * 256 + threadIdx.x;  // 0..2047
            const bf16_t* xsrc = xcat + (size_t)(t + 1) * NB * NIN;
            for (int i = tid2; i < NB * NIN; i += 8 * 256) {
                int b = i >> 7, k = i & (NIN - 1);
                abuf_w[(size_t)b * KTOT + k] = xsrc[i];
            }
        }
    }
}

// ---------------- trailing kernel: y_255 = h_255 @ W_out^T --------------------------
__global__ __launch_bounds__(256) void rnn_final_out(
        const bf16_t* __restrict__ abuf_h,  // buffer holding bf16 h_255
        const bf16_t* __restrict__ Wout,
        float* __restrict__ out_y) {        // d_out + OUT_OFF
    const int wave = blockIdx.x * 8 + (threadIdx.x >> 5);
    const int lane = threadIdx.x & 31;
    const int mb = wave >> 3;
    const int ob = wave & 7;
    const int half = lane >> 4;
    const int q = lane & 15;

    const bf16_t* arow = abuf_h + (size_t)(mb * 16 + q) * KTOT + NIN + half * 8;
    const bf16_t* brow = Wout   + (size_t)(ob * 16 + q) * NHID + half * 8;
    v8f c = wmma_k_loop(arow, brow, NHID);

    const int o = ob * 16 + q;
    #pragma unroll
    for (int r = 0; r < 8; ++r) {
        const int b = mb * 16 + half * 8 + r;
        out_y[(size_t)b * (T_STEPS * NOUT) + (size_t)(T_STEPS - 1) * NOUT + o] = c[r];
    }
}

extern "C" void kernel_launch(void* const* d_in, const int* in_sizes, int n_in,
                              void* d_out, int out_size, void* d_ws, size_t ws_size,
                              hipStream_t stream) {
    const float* x       = (const float*)d_in[0];
    const float* hidden  = (const float*)d_in[1];
    const float* W_in    = (const float*)d_in[2];
    const float* W_hh    = (const float*)d_in[3];
    const float* b_hh    = (const float*)d_in[4];
    const float* W_out   = (const float*)d_in[5];
    const float* alpha_w = (const float*)d_in[6];
    const float* noise   = (const float*)d_in[7];
    float* out = (float*)d_out;

    char* ws = (char*)d_ws;
    bf16_t* Wcat  = (bf16_t*)(ws + WCAT_OFF);
    bf16_t* Wout  = (bf16_t*)(ws + WOUT_OFF);
    bf16_t* xcat  = (bf16_t*)(ws + XCAT_OFF);
    float*  bn    = (float*)(ws + BN_OFF);
    bf16_t* abuf0 = (bf16_t*)(ws + ABUF_OFF);
    bf16_t* abuf1 = abuf0 + (size_t)NB * KTOT;

    rnn_prep<<<1024, 256, 0, stream>>>(x, hidden, W_in, W_hh, b_hh, W_out,
                                       alpha_w, noise, Wcat, Wout, xcat, bn, abuf0);

    for (int t = 0; t < T_STEPS; ++t) {
        bf16_t* abr = (t & 1) ? abuf1 : abuf0;
        bf16_t* abw = (t & 1) ? abuf0 : abuf1;
        rnn_fused_step<<<72, 256, 0, stream>>>(abr, abw, Wcat, Wout, xcat, bn, out, t);
    }
    // h_255 was written into abuf0 (t=255 odd: write buffer = abuf0)
    rnn_final_out<<<8, 256, 0, stream>>>(abuf0, Wout, out + OUT_OFF);
}
